// LSTM_ReturnAllChunk_64639257805417
// MI455X (gfx1250) — matrix operand order repaired and checked by reading, not compiled, not run on hardware
//
#include <hip/hip_runtime.h>
#include <math.h>

// Problem constants (from reference): BS=1024, T=8, D=512, H=512
#define NROWS   8192          // BS*T
#define DIN     512
#define HID     512
#define FOURH   2048          // 4*H

// Scan decomposition: 16 WGPs per direction, each owns 32 h-elements
#define NWG     16
#define SLICE   32            // h elements per workgroup
#define SROWS   128           // W_hh rows per workgroup (4 gates x SLICE)

typedef __attribute__((ext_vector_type(16))) __bf16 v16bf;
typedef __attribute__((ext_vector_type(8)))  __bf16 v8bf;
typedef __attribute__((ext_vector_type(8)))  float  v8f;

// ---------------------------------------------------------------------------
// fp32 -> bf16 conversion (for WMMA A/B operands)
// ---------------------------------------------------------------------------
__global__ void f32_to_bf16_kernel(const float* __restrict__ src,
                                   __bf16* __restrict__ dst, int n) {
  int i = blockIdx.x * blockDim.x + threadIdx.x;
  if (i < n) dst[i] = (__bf16)src[i];
}

// ---------------------------------------------------------------------------
// Gate GEMM: G[n, j] = sum_k X[rowmap(n), k] * W[j, k] + bias[j]
// One wave computes one 16x16 tile via v_wmma_f32_16x16x32_bf16 (K=512 -> 16
// WMMA steps). dir==1 remaps rows for the per-sample-reversed backward pass.
// ---------------------------------------------------------------------------
__global__ __launch_bounds__(256)
void lstm_gate_gemm_wmma(const __bf16* __restrict__ X,    // NROWS x DIN
                         const __bf16* __restrict__ W,    // FOURH x DIN
                         const float*  __restrict__ bias, // FOURH
                         float*        __restrict__ G,    // NROWS x FOURH
                         int dir) {
  const int lane = threadIdx.x & 31;
  const int wave = threadIdx.x >> 5;
  const int col_tile = blockIdx.x * 8 + wave;   // 0..127
  const int row_tile = blockIdx.y;              // 0..511
  const int half = lane >> 4;
  const int l16  = lane & 15;

  int n = row_tile * 16 + l16;
  int r = n;
  if (dir) { r = (n & ~7) + (7 - (n & 7)); }    // per-sample time reversal (T=8)
  const __bf16* __restrict__ arow = X + (size_t)r * DIN;
  const int j = col_tile * 16 + l16;
  const __bf16* __restrict__ bcol = W + (size_t)j * DIN;

  v8f acc = {};
  #pragma unroll 4
  for (int kk = 0; kk < DIN; kk += 32) {
    v16bf a = *(const v16bf*)(arow + kk + half * 16);
    v16bf b = *(const v16bf*)(bcol + kk + half * 16);
    acc = __builtin_amdgcn_wmma_f32_16x16x32_bf16(
        false, a, false, b, (short)0, acc, false, false);
  }

  const float bv = bias[j];
  #pragma unroll
  for (int v = 0; v < 8; ++v) {
    const int m = row_tile * 16 + v + half * 8;
    G[(size_t)m * FOURH + j] = acc[v] + bv;
  }
}

// ---------------------------------------------------------------------------
// Scan init: reset barrier counters and seed the h exchange buffers (runs
// inside the graph so every replay is deterministic).
// ---------------------------------------------------------------------------
__global__ void scan_init(const float* __restrict__ h0,
                          float* __restrict__ hbuf,     // [2][2][HID]
                          unsigned* __restrict__ ctr) { // [2]
  int i = blockIdx.x * blockDim.x + threadIdx.x;
  if (i < 2) ctr[i] = 0u;
  if (i < 2 * HID) {
    int dir = i >> 9, j = i & (HID - 1);
    hbuf[dir * (2 * HID) + j] = h0[i];   // buffer 0 = input state of step 0
  }
}

// ---------------------------------------------------------------------------
// Cooperative LSTM scan: 16 workgroups per direction (one per WGP), each
// holding its 128 W_hh rows (256 KB fp32) resident in the 320 KB LDS for the
// whole 8192-step scan. Steps are synchronized with a monotonic L2 atomic
// barrier; h state is exchanged through a double-buffered L2 vector.
// W_hh staging uses CDNA5 async global->LDS copies (ASYNCcnt path).
// ---------------------------------------------------------------------------
__global__ __launch_bounds__(SROWS)
void lstm_scan_mw(const float* __restrict__ Gf, const float* __restrict__ Gb,
                  const float* __restrict__ Whf, const float* __restrict__ Whb,
                  const float* __restrict__ c0,
                  float* __restrict__ hbuf,      // [2][2][HID]
                  unsigned* __restrict__ ctr,    // [2]
                  float* __restrict__ out,       // NROWS x (2*HID)
                  float* __restrict__ hc_out) {  // h_n (2x512) then c_n (2x512)
  extern __shared__ char smem[];                 // dynamic LDS starts at offset 0
  float* Wl    = (float*)smem;                           // SROWS x 512 fp32
  float* hfull = (float*)(smem + SROWS * DIN * 4);       // 512 floats
  float* zloc  = (float*)(smem + SROWS * DIN * 4 + 2048);// SROWS floats

  const int dir = blockIdx.y;
  const int wg  = blockIdx.x;
  const int tid = threadIdx.x;
  const int s   = wg * SLICE;

  const float* __restrict__ G = dir ? Gb : Gf;
  const float* __restrict__ W = dir ? Whb : Whf;
  float*    hb    = hbuf + dir * (2 * HID);
  unsigned* c_ptr = ctr + dir;

  // ---- Stage this WG's 128 W_hh rows into LDS (async global->LDS B128) ----
  // Each thread copies 16 bytes per row: LDS byte offset = lr*2048 + tid*16.
  for (int lr = 0; lr < SROWS; ++lr) {
    const int jr = ((lr >> 5) << 9) + s + (lr & 31);  // gate*512 + s + idx
    const unsigned lds_off = (unsigned)(lr * 2048 + tid * 16);
    const float* gp = W + (size_t)jr * DIN + tid * 4;
    asm volatile("global_load_async_to_lds_b128 %0, %1, off"
                 :: "v"(lds_off), "v"(gp) : "memory");
  }
#if __has_builtin(__builtin_amdgcn_s_wait_asynccnt)
  __builtin_amdgcn_s_wait_asynccnt(0);
#else
  asm volatile("s_wait_asynccnt 0" ::: "memory");
#endif
  __syncthreads();

  // Local cell state: thread t (<32) owns h/c element s+t of this direction.
  float c_reg = 0.0f;
  if (tid < SLICE) c_reg = c0[dir * HID + s + tid];

  const int  myrow = tid;                               // 0..127
  const int  jr    = ((myrow >> 5) << 9) + s + (myrow & 31);
  const float* __restrict__ wrow = Wl + myrow * DIN;

  for (int n = 0; n < NROWS; ++n) {
    // Pull current h (512 floats) from the exchange buffer into LDS.
    {
      const float4 hv4 = *(const float4*)(hb + (n & 1) * HID + tid * 4);
      *(float4*)(hfull + tid * 4) = hv4;
    }
    __syncthreads();

    // z[jr] = g[n, jr] + dot(h, W_hh[jr, :])  -- one row per thread, all LDS.
    const float* __restrict__ g = G + (size_t)n * FOURH;
    if (n + 1 < NROWS) __builtin_prefetch(g + FOURH + jr, 0, 0);
    float sacc = g[jr];
    #pragma unroll 8
    for (int k4 = 0; k4 < DIN / 4; ++k4) {
      const float4 wv = *(const float4*)(wrow + k4 * 4);
      const float4 hv = *(const float4*)(hfull + k4 * 4);
      sacc = fmaf(wv.x, hv.x, sacc);
      sacc = fmaf(wv.y, hv.y, sacc);
      sacc = fmaf(wv.z, hv.z, sacc);
      sacc = fmaf(wv.w, hv.w, sacc);
    }
    zloc[myrow] = sacc;
    __syncthreads();

    if (tid < SLICE) {
      // PyTorch gate order: i, f, g, o (rows 0..31, 32..63, 64..95, 96..127)
      const float iv = 1.0f / (1.0f + __expf(-zloc[tid]));
      const float fv = 1.0f / (1.0f + __expf(-zloc[SLICE + tid]));
      const float gv = tanhf(zloc[2 * SLICE + tid]);
      const float ov = 1.0f / (1.0f + __expf(-zloc[3 * SLICE + tid]));
      c_reg = fv * c_reg + iv * gv;
      const float hv = ov * tanhf(c_reg);
      hb[((n + 1) & 1) * HID + s + tid] = hv;           // next-step buffer
      int outrow = dir ? ((n & ~7) + (7 - (n & 7))) : n;
      out[(size_t)outrow * (2 * HID) + dir * HID + s + tid] = hv;
    }

    // Inter-WGP step barrier: monotonic counter, release add / acquire spin.
    __syncthreads();
    if (tid == 0) {
      __hip_atomic_fetch_add(c_ptr, 1u, __ATOMIC_RELEASE, __HIP_MEMORY_SCOPE_AGENT);
      const unsigned target = (unsigned)(n + 1) * NWG;
      while (__hip_atomic_load(c_ptr, __ATOMIC_ACQUIRE, __HIP_MEMORY_SCOPE_AGENT) < target) {
        __builtin_amdgcn_s_sleep(1);
      }
    }
    __syncthreads();
  }

  if (tid < SLICE) {
    hc_out[dir * HID + s + tid] = hb[(NROWS & 1) * HID + s + tid]; // h_n
    hc_out[2 * HID + dir * HID + s + tid] = c_reg;                 // c_n
  }
}

// ---------------------------------------------------------------------------
// Host launcher
// ---------------------------------------------------------------------------
extern "C" void kernel_launch(void* const* d_in, const int* in_sizes, int n_in,
                              void* d_out, int out_size, void* d_ws, size_t ws_size,
                              hipStream_t stream) {
  const float* x     = (const float*)d_in[0];
  const float* h0    = (const float*)d_in[1];
  const float* c0    = (const float*)d_in[2];
  const float* W_ihf = (const float*)d_in[3];
  const float* W_hhf = (const float*)d_in[4];
  const float* b_f   = (const float*)d_in[5];
  const float* W_ihb = (const float*)d_in[6];
  const float* W_hhb = (const float*)d_in[7];
  const float* b_b   = (const float*)d_in[8];

  float* out = (float*)d_out;
  float* hc_out = out + (size_t)NROWS * 2 * HID;

  // Workspace carve-up (~140 MB + 16 KB)
  char* ws = (char*)d_ws;
  __bf16*   xbf  = (__bf16*)(ws);                       //  8 MB
  __bf16*   wfb  = (__bf16*)(ws + ((size_t)8  << 20));  //  2 MB
  __bf16*   wbb  = (__bf16*)(ws + ((size_t)10 << 20));  //  2 MB
  float*    Gf   = (float*) (ws + ((size_t)12 << 20));  // 64 MB
  float*    Gb   = (float*) (ws + ((size_t)76 << 20));  // 64 MB
  float*    hbuf = (float*) (ws + ((size_t)140 << 20)); //  8 KB: [2][2][512]
  unsigned* ctr  = (unsigned*)(ws + ((size_t)140 << 20) + 8192);

  // 1) fp32 -> bf16 conversions
  {
    const int nx = NROWS * DIN;
    const int nw = FOURH * DIN;
    f32_to_bf16_kernel<<<nx / 256, 256, 0, stream>>>(x, xbf, nx);
    f32_to_bf16_kernel<<<nw / 256, 256, 0, stream>>>(W_ihf, wfb, nw);
    f32_to_bf16_kernel<<<nw / 256, 256, 0, stream>>>(W_ihb, wbb, nw);
  }

  // 2) Reset barrier counters + seed h exchange buffers (in-graph, per call)
  scan_init<<<4, 256, 0, stream>>>(h0, hbuf, ctr);

  // 3) Gate GEMMs (WMMA bf16, fp32 accumulate)
  {
    dim3 grid(FOURH / (16 * 8), NROWS / 16); // (16, 512)
    lstm_gate_gemm_wmma<<<grid, 256, 0, stream>>>(xbf, wfb, b_f, Gf, 0);
    lstm_gate_gemm_wmma<<<grid, 256, 0, stream>>>(xbf, wbb, b_b, Gb, 1);
  }

  // 4) Cooperative scan: 16 WGPs per direction, LDS-resident W_hh
  {
    dim3 grid(NWG, 2);
    const size_t lds_bytes = (size_t)SROWS * DIN * 4 + 2048 + SROWS * 4; // 264,704 B
    lstm_scan_mw<<<grid, SROWS, lds_bytes, stream>>>(
        Gf, Gb, W_hhf, W_hhb, c0, hbuf, ctr, out, hc_out);
  }
}
